// SirenModel_70927089926739
// MI455X (gfx1250) — compile-verified, matrix operand order checked by reading, not run on hardware
//
#include <hip/hip_runtime.h>
#include <hip/hip_bf16.h>

// ---------------------------------------------------------------------------
// SIREN hypernetwork forward on MI455X (gfx1250).
// Compute-bound (137 GFLOP vs ~22MB HBM) -> bf16 WMMA (v_wmma_f32_16x16x32_bf16)
// with f32 accumulation. Per-(sample, 128-row) workgroup keeps activations and
// the per-sample 256x256 weight matrix resident in CDNA5's 320KB LDS.
// ---------------------------------------------------------------------------

typedef __attribute__((ext_vector_type(16))) __bf16 v16bf;
typedef __attribute__((ext_vector_type(8)))  float  v8f;

#define NC      16
#define HWC     16384
#define CIC     2
#define DIMC    256
#define DEPTHC  4
#define COC     3
#define OMEGA_C 30.0f
#define E_C     263936           // CI*DIM + DIM + DEPTH*(DIM*DIM+DIM)

#define ROWS    128              // HW rows per workgroup
#define STRIDE  264              // padded LDS row stride (bf16 elems): breaks bank conflicts
#define WT_BYTES   (DIMC * STRIDE * 2)            // 135168
#define H_BYTES    (ROWS * STRIDE * 2)            // 67584
#define LDS_BYTES  (WT_BYTES + 2 * H_BYTES + DIMC * 4)  // 271360 <= 320KB

struct F32x8 { uint4 lo; uint4 hi; };             // 32B == v16bf

union Pack8 { __bf16 h[8]; uint4 v; };            // 16B packed bf16 store

// Load one v16bf WMMA fragment: two contiguous 8-element (16B) k-runs.
// ISA 7.12.2 (16-bit A 16x32): lane L holds row/col L%16; with g = L/16,
// vector elems 0..7 = K kb+8g+{0..7}, elems 8..15 = K kb+16+8g+{0..7}.
__device__ __forceinline__ v16bf ld_frag(const __bf16* p0, const __bf16* p1) {
    F32x8 t;
    t.lo = *(const uint4*)p0;
    t.hi = *(const uint4*)p1;
    return __builtin_bit_cast(v16bf, t);
}

__global__ void __launch_bounds__(256, 1)
siren_wmma_kernel(const float* __restrict__ x,      // (N, HW, CI)
                  const float* __restrict__ woff,   // (N, E)
                  const float* __restrict__ w0,     // (CI, DIM)
                  const float* __restrict__ b0,     // (DIM)
                  const float* __restrict__ ws,     // (DEPTH, DIM, DIM)
                  const float* __restrict__ bs,     // (DEPTH, DIM)
                  const float* __restrict__ wout,   // (DIM, CO)
                  const float* __restrict__ bout,   // (CO)
                  const float* __restrict__ oscale, // (CO)
                  float* __restrict__ out)          // (N, HW, CO)
{
    extern __shared__ __align__(16) char smem[];
    __bf16* WT = (__bf16*)smem;                               // 256 x 264 (W^T, bf16)
    __bf16* H0 = (__bf16*)(smem + WT_BYTES);                  // 128 x 264
    __bf16* H1 = (__bf16*)(smem + WT_BYTES + H_BYTES);        // 128 x 264
    float*  BV = (float*)(smem + WT_BYTES + 2 * H_BYTES);     // fused bias, 256 f32

    const int tid  = threadIdx.x;
    const int lane = tid & 31;
    const int wave = tid >> 5;          // 8 wave32 per workgroup
    const int samp = blockIdx.y;
    const int row0 = blockIdx.x * ROWS;

    const float* wo = woff + (size_t)samp * E_C;   // this sample's offsets

    // ---------------- Layer 0: h = sin(OMEGA * (x @ (w0+ow0) + (b0+ob0))) ---
    {
        const int d = tid;                                  // output column
        const float w00 = w0[d]        + wo[d];             // ow0[0][d]
        const float w01 = w0[DIMC + d] + wo[DIMC + d];      // ow0[1][d]
        const float bb  = b0[d]        + wo[2 * DIMC + d];  // ob0[d]
        const float* xb = x + ((size_t)samp * HWC + row0) * CIC;
        for (int r = 0; r < ROWS; ++r) {
            float x0 = xb[2 * r + 0];
            float x1 = xb[2 * r + 1];
            float h  = __sinf(OMEGA_C * (x0 * w00 + x1 * w01 + bb));
            H0[r * STRIDE + d] = (__bf16)h;
        }
    }
    __syncthreads();

    int cur = 0;                                            // H0 holds current h
    // ---------------- Hidden layers: h = sin(h @ (ws[l]+ow) + (bs[l]+ob)) ---
    for (int l = 0; l < DEPTHC; ++l) {
        const float* wsl = ws + (size_t)l * DIMC * DIMC;
        const float* owl = wo + 3 * DIMC + (size_t)l * (DIMC * DIMC + DIMC);

        BV[tid] = bs[l * DIMC + tid] + owl[DIMC * DIMC + tid];

        // Build W^T bf16 in LDS: WT[n][k] = wsl[k][n] + owl[k][n].
        // Wave w owns k-block [32w,32w+32); lane -> n, so global reads are
        // fully coalesced (consecutive n) and LDS writes are 16B vectors.
        const int kb0 = wave * 32;
        for (int it = 0; it < 8; ++it) {
            const int n = it * 32 + lane;
            __bf16* dst = WT + n * STRIDE + kb0;
            #pragma unroll
            for (int kc = 0; kc < 4; ++kc) {
                Pack8 p;
                #pragma unroll
                for (int j = 0; j < 8; ++j) {
                    const int k = kb0 + kc * 8 + j;
                    p.h[j] = (__bf16)(wsl[k * DIMC + n] + owl[k * DIMC + n]);
                }
                *(uint4*)(dst + kc * 8) = p.v;
            }
        }
        __syncthreads();

        const __bf16* Hc = cur ? H1 : H0;
        __bf16*       Hn = cur ? H0 : H1;
        const int m  = lane & 15;
        const int g  = lane >> 4;
        const int rb = wave;                  // wave w -> rows [16w, 16w+16)

        // Hoist the 8 A fragments (full K) for this wave's row block.
        const __bf16* rowp = Hc + (rb * 16 + m) * STRIDE;
        v16bf a[8];
        #pragma unroll
        for (int kb = 0; kb < 8; ++kb)
            a[kb] = ld_frag(rowp + kb * 32 + 8 * g,
                            rowp + kb * 32 + 16 + 8 * g);

        for (int cb = 0; cb < 16; ++cb) {     // 16 column tiles of 16
            const __bf16* colp = WT + (cb * 16 + m) * STRIDE;
            v8f acc = {};
            #pragma unroll
            for (int kb = 0; kb < 8; ++kb) {
                v16bf b = ld_frag(colp + kb * 32 + 8 * g,
                                  colp + kb * 32 + 16 + 8 * g);
                acc = __builtin_amdgcn_wmma_f32_16x16x32_bf16(
                          false, a[kb], false, b, (short)0, acc, false, false);
            }
            const float bv = BV[cb * 16 + m];
            // C/D layout: acc[i] -> row rb*16 + i + 8g, col cb*16 + m.
            #pragma unroll
            for (int i = 0; i < 8; ++i) {
                float v = __sinf(acc[i] + bv);
                Hn[(rb * 16 + i + 8 * g) * STRIDE + cb * 16 + m] = (__bf16)v;
            }
        }
        cur ^= 1;
        __syncthreads();   // Hn complete; safe to rebuild WT/BV next iter
    }

    // ---------------- Output: (h @ w_out + b_out) * out_scale (CO=3) -------
    const __bf16* Hf = cur ? H1 : H0;
    if (tid < ROWS) {
        const int r = tid;
        float a0 = 0.f, a1 = 0.f, a2 = 0.f;
        for (int k = 0; k < DIMC; ++k) {
            const float h = (float)Hf[r * STRIDE + k];
            a0 += h * wout[k * COC + 0];
            a1 += h * wout[k * COC + 1];
            a2 += h * wout[k * COC + 2];
        }
        float* op = out + ((size_t)samp * HWC + row0 + r) * COC;
        op[0] = (a0 + bout[0]) * oscale[0];
        op[1] = (a1 + bout[1]) * oscale[1];
        op[2] = (a2 + bout[2]) * oscale[2];
    }
}

extern "C" void kernel_launch(void* const* d_in, const int* in_sizes, int n_in,
                              void* d_out, int out_size, void* d_ws, size_t ws_size,
                              hipStream_t stream) {
    const float* x      = (const float*)d_in[0];
    const float* woff   = (const float*)d_in[1];
    const float* w0     = (const float*)d_in[2];
    const float* b0     = (const float*)d_in[3];
    const float* ws     = (const float*)d_in[4];
    const float* bs     = (const float*)d_in[5];
    const float* wout   = (const float*)d_in[6];
    const float* bout   = (const float*)d_in[7];
    const float* oscale = (const float*)d_in[8];
    float* out = (float*)d_out;

    // Allow 271KB dynamic LDS per workgroup (CDNA5 WGP: up to 320KB).
    (void)hipFuncSetAttribute((const void*)siren_wmma_kernel,
                              hipFuncAttributeMaxDynamicSharedMemorySize,
                              (int)LDS_BYTES);

    dim3 grid(HWC / ROWS, NC);   // 128 row-chunks x 16 samples
    siren_wmma_kernel<<<grid, 256, LDS_BYTES, stream>>>(
        x, woff, w0, b0, ws, bs, wout, bout, oscale, out);
}
